// GATLayoutModel_51848845197431
// MI455X (gfx1250) — matrix operand order repaired and checked by reading, not compiled, not run on hardware
//
#include <hip/hip_runtime.h>
#include <hip/hip_bf16.h>
#include <stdint.h>

#define BB   32
#define NN   10000
#define EE   40000
#define NCC  1000
#define CC   128
#define HH   16
#define KCAT 256        // 234 padded to a multiple of 32
#define BN   (BB * NN)  // 320000 rows, exactly 20000 16-row tiles

typedef uint32_t u32;
typedef __attribute__((ext_vector_type(8)))  u32    v8u;
typedef __attribute__((ext_vector_type(8)))  float  v8f;
typedef __attribute__((ext_vector_type(16))) __bf16 v16bf;

// ---------------- bf16 helpers (round-to-nearest-even) ----------------
__device__ __forceinline__ u32 f2bf(float f) {
  u32 u = __builtin_bit_cast(u32, f);
  u32 r = u + 0x7FFFu + ((u >> 16) & 1u);
  return r >> 16;
}
__device__ __forceinline__ u32 pkbf(float a, float b) {
  return f2bf(a) | (f2bf(b) << 16);
}
__device__ __forceinline__ v16bf as_bf(v8u a) {
  return __builtin_bit_cast(v16bf, a);
}

// A operand, 16x32 bf16 tile from a bf16 row-major source.
// lanes 0-15: m=lane, K = {kb..kb+7} in v0-3 and {kb+16..kb+23} in v4-7, kb=0
// lanes 16-31: same rows, kb=8.
__device__ __forceinline__ v16bf load_a_bf16(const uint16_t* rowp, int k0) {
  const uint4* p0 = (const uint4*)(rowp + k0);
  const uint4* p1 = (const uint4*)(rowp + k0 + 16);
  uint4 x0 = *p0, x1 = *p1;
  v8u a;
  a[0] = x0.x; a[1] = x0.y; a[2] = x0.z; a[3] = x0.w;
  a[4] = x1.x; a[5] = x1.y; a[6] = x1.z; a[7] = x1.w;
  return as_bf(a);
}

// Same A layout, but source is f32: load 2x8 floats, pack to bf16 in-register.
__device__ __forceinline__ v16bf load_a_f32(const float* rowp, int k0) {
  const float4* p = (const float4*)(rowp + k0);
  float4 f0 = p[0], f1 = p[1];
  const float4* q = (const float4*)(rowp + k0 + 16);
  float4 f2 = q[0], f3 = q[1];
  v8u a;
  a[0] = pkbf(f0.x, f0.y); a[1] = pkbf(f0.z, f0.w);
  a[2] = pkbf(f1.x, f1.y); a[3] = pkbf(f1.z, f1.w);
  a[4] = pkbf(f2.x, f2.y); a[5] = pkbf(f2.z, f2.w);
  a[6] = pkbf(f3.x, f3.y); a[7] = pkbf(f3.z, f3.w);
  return as_bf(a);
}

// B operand from the pre-packed weight image: 8 consecutive dwords per lane.
__device__ __forceinline__ v16bf load_b(const u32* pk, int kstep, int ntile, int lane) {
  const uint4* q = (const uint4*)(pk + ((((kstep * 8 + ntile) * 32) + lane) << 3));
  uint4 b0 = q[0], b1 = q[1];
  v8u b;
  b[0] = b0.x; b[1] = b0.y; b[2] = b0.z; b[3] = b0.w;
  b[4] = b1.x; b[5] = b1.y; b[6] = b1.z; b[7] = b1.w;
  return as_bf(b);
}

__device__ __forceinline__ v8f wmma_bf16(v16bf a, v16bf b, v8f c) {
  return __builtin_amdgcn_wmma_f32_16x16x32_bf16(false, a, false, b, (short)0, c, false, false);
}

// ---------------- setup kernels ----------------
__global__ void k_init_misc(int* __restrict__ cfg_pos, float* __restrict__ deg) {
  int i = blockIdx.x * blockDim.x + threadIdx.x;
  if (i < NN) { cfg_pos[i] = -1; deg[i] = 0.f; }
}

__global__ void k_scatter_cfg(const int* __restrict__ ids, int* __restrict__ cfg_pos) {
  int i = blockIdx.x * blockDim.x + threadIdx.x;
  if (i < NCC) cfg_pos[ids[i]] = i;
}

__global__ void k_deg(const int* __restrict__ ei, float* __restrict__ deg) {
  int e = blockIdx.x * blockDim.x + threadIdx.x;
  if (e < EE) atomicAdd(&deg[ei[EE + e]], 1.f);
}

__global__ void k_invdeg(float* __restrict__ deg) {
  int i = blockIdx.x * blockDim.x + threadIdx.x;
  if (i < NN) deg[i] = 1.f / fmaxf(deg[i], 1.f);
}

// Pack W[K, 128] (f32 row-major) into the WMMA B-operand image:
// idx = ((ktile*8 + ntile)*32 + lane)*8 + v ; lanes 0-15: v0-3 K=0..7, v4-7 K=16..23,
// lanes 16-31: v0-3 K=8..15, v4-7 K=24..31 ; n = ntile*16 + (lane&15). Zero-pads K >= Kreal.
__global__ void k_pack_w(const float* __restrict__ W, u32* __restrict__ P,
                         int Kreal, int total) {
  int idx = blockIdx.x * blockDim.x + threadIdx.x;
  if (idx >= total) return;
  int v     = idx & 7;
  int lane  = (idx >> 3) & 31;
  int ntile = (idx >> 8) & 7;
  int ktile = idx >> 11;
  int n  = ntile * 16 + (lane & 15);
  int k0 = ktile * 32 + (lane >> 4) * 8 + 2 * v + (v & 4) * 2;
  float a = (k0     < Kreal) ? W[k0 * CC + n]       : 0.f;
  float b = (k0 + 1 < Kreal) ? W[(k0 + 1) * CC + n] : 0.f;
  P[idx] = pkbf(a, b);
}

// ---------------- feature concat (bf16, K padded to 256) ----------------
__global__ __launch_bounds__(256) void k_build_xcat(
    const int* __restrict__ xcfg, const float* __restrict__ xfeat,
    const int* __restrict__ xlay, const int* __restrict__ xop,
    const int* __restrict__ cfg_pos, const float* __restrict__ emb_op,
    const float* __restrict__ emb_cfg, uint16_t* __restrict__ Xcat) {
  long row = blockIdx.x;
  int  c   = threadIdx.x;
  int  b   = (int)(row / NN);
  int  n   = (int)(row - (long)b * NN);
  float v = 0.f;
  if (c < 134) {
    v = xfeat[(long)n * 134 + c];
  } else if (c < 206) {          // 18 config embeddings x 4
    int k = c - 134, slot = k >> 2, comp = k & 3;
    int pos = cfg_pos[n];
    int idx = (pos >= 0) ? xcfg[((long)b * NCC + pos) * 18 + slot] : -1;
    v = emb_cfg[(idx + 1) * 4 + comp];
  } else if (c < 230) {          // 6 layout embeddings x 4
    int k = c - 206, slot = k >> 2, comp = k & 3;
    v = emb_cfg[(xlay[n * 6 + slot] + 1) * 4 + comp];
  } else if (c < 234) {          // op embedding x 4
    v = emb_op[xop[n] * 4 + (c - 230)];
  }
  Xcat[row * KCAT + c] = (uint16_t)f2bf(v);
}

// ---------------- WMMA GEMMs ----------------
// Y[320000,128] = Xcat(bf16)[.,256] @ Wpk + bias. 8 waves/block: wave=ntile.
__global__ __launch_bounds__(256) void k_gemm_lin0(
    const uint16_t* __restrict__ Xcat, const u32* __restrict__ Wpk,
    const float* __restrict__ bias, float* __restrict__ Y) {
  int lane  = threadIdx.x & 31;
  int ntile = threadIdx.x >> 5;
  long rowbase = (long)blockIdx.x * 16;
  int m  = lane & 15;
  int kb = (lane >> 4) * 8;
  const uint16_t* arow = Xcat + (rowbase + m) * KCAT;
  v8f acc = {};
#pragma unroll
  for (int ks = 0; ks < KCAT / 32; ++ks) {
    v16bf a = load_a_bf16(arow, ks * 32 + kb);
    v16bf b = load_b(Wpk, ks, ntile, lane);
    acc = wmma_bf16(a, b, acc);
  }
  int col = ntile * 16 + m;
  float bv = bias[col];
  int rsel = (lane >> 4) * 8;   // C layout: vgpr v -> M=v (lanes 0-15) / v+8
#pragma unroll
  for (int v = 0; v < 8; ++v) Y[(rowbase + rsel + v) * CC + col] = acc[v] + bv;
}

// Y = bf16(Agg) @ WL + bf16(H) @ WR + bias, K = 128 each.
__global__ __launch_bounds__(256) void k_gemm_conv(
    const float* __restrict__ Agg, const float* __restrict__ H,
    const u32* __restrict__ WLpk, const u32* __restrict__ WRpk,
    const float* __restrict__ bias, float* __restrict__ Y) {
  int lane  = threadIdx.x & 31;
  int ntile = threadIdx.x >> 5;
  long rowbase = (long)blockIdx.x * 16;
  int m  = lane & 15;
  int kb = (lane >> 4) * 8;
  const float* arow1 = Agg + (rowbase + m) * CC;
  const float* arow2 = H   + (rowbase + m) * CC;
  v8f acc = {};
#pragma unroll
  for (int ks = 0; ks < CC / 32; ++ks)
    acc = wmma_bf16(load_a_f32(arow1, ks * 32 + kb), load_b(WLpk, ks, ntile, lane), acc);
#pragma unroll
  for (int ks = 0; ks < CC / 32; ++ks)
    acc = wmma_bf16(load_a_f32(arow2, ks * 32 + kb), load_b(WRpk, ks, ntile, lane), acc);
  int col = ntile * 16 + m;
  float bv = bias[col];
  int rsel = (lane >> 4) * 8;
#pragma unroll
  for (int v = 0; v < 8; ++v) Y[(rowbase + rsel + v) * CC + col] = acc[v] + bv;
}

// ---------------- instance-norm stats over node axis, per (b, c) ----------------
__global__ __launch_bounds__(256) void k_stats(const float* __restrict__ Y,
                                               float* __restrict__ stats) {
  __shared__ float ssum[256], ssq[256];
  int b = blockIdx.x;
  int c = threadIdx.x & 127;
  int half = threadIdx.x >> 7;
  float sum = 0.f, sq = 0.f;
  const float* p = Y + ((long)b * NN) * CC + c;
  for (int n = half * (NN / 2); n < (half + 1) * (NN / 2); ++n) {
    float x = p[(long)n * CC];
    sum += x; sq += x * x;
  }
  ssum[threadIdx.x] = sum; ssq[threadIdx.x] = sq;
  __syncthreads();
  if (half == 0) {
    sum += ssum[threadIdx.x + 128];
    sq  += ssq[threadIdx.x + 128];
    float mean = sum * (1.f / NN);
    float var  = sq * (1.f / NN) - mean * mean;
    stats[b * 2 * CC + c]      = mean;
    stats[b * 2 * CC + CC + c] = rsqrtf(var + 1e-5f);
  }
}

// ---------------- fused norm + channel-attention + relu ----------------
// mode 0 : h = relu(attn(instnorm(Y)))                  (after lin0)
// mode 1 : h = relu(instnorm(Y) + attn(Hold))           (conv layers; in-place ok)
__global__ __launch_bounds__(128) void k_norm_attn(
    const float* __restrict__ Y, const float* __restrict__ stats,
    const float* __restrict__ Hold,
    const float* __restrict__ w1, const float* __restrict__ b1,
    const float* __restrict__ w2, const float* __restrict__ b2,
    float* __restrict__ Hout, int mode) {
  __shared__ float xv[CC];
  __shared__ float t[HH];
  long row = blockIdx.x;
  int c = threadIdx.x;
  int b = (int)(row / NN);
  float mean = stats[b * 2 * CC + c];
  float rstd = stats[b * 2 * CC + CC + c];
  float normed = (Y[row * CC + c] - mean) * rstd;
  float asrc = (mode == 0) ? normed : Hold[row * CC + c];
  xv[c] = asrc;
  __syncthreads();
  if (c < HH) {
    float s = b1[c];
#pragma unroll 8
    for (int k = 0; k < CC; ++k) s += xv[k] * w1[k * HH + c];
    t[c] = fmaxf(s, 0.f);
  }
  __syncthreads();
  float s2 = b2[c];
#pragma unroll
  for (int j = 0; j < HH; ++j) s2 += t[j] * w2[j * CC + c];
  float attn = asrc / (1.f + __expf(-s2));   // sigmoid(s2) * asrc
  float outv = (mode == 0) ? fmaxf(attn, 0.f) : fmaxf(normed + attn, 0.f);
  Hout[row * CC + c] = outv;
}

// ---------------- graph scatter-mean ----------------
__global__ void k_zero(float* __restrict__ p, long n) {
  long i = (long)blockIdx.x * blockDim.x + threadIdx.x;
  if (i < n) p[i] = 0.f;
}

__global__ __launch_bounds__(256) void k_scatter(
    const int* __restrict__ ei, const float* __restrict__ inv_deg,
    const float* __restrict__ H, float* __restrict__ Agg) {
  int e = blockIdx.x;
  int s = ei[e], d = ei[EE + e];
  float w = inv_deg[d];
  for (int i = threadIdx.x; i < BB * CC; i += 256) {
    int b = i >> 7, c = i & 127;
    atomicAdd(&Agg[((long)b * NN + d) * CC + c],
              H[((long)b * NN + s) * CC + c] * w);
  }
}

// ---------------- global mean pool ----------------
__global__ __launch_bounds__(256) void k_pool(const float* __restrict__ H,
                                              float* __restrict__ G) {
  __shared__ float ssum[256];
  int b = blockIdx.x;
  int c = threadIdx.x & 127;
  int half = threadIdx.x >> 7;
  float sum = 0.f;
  const float* p = H + ((long)b * NN) * CC + c;
  for (int n = half * (NN / 2); n < (half + 1) * (NN / 2); ++n) sum += p[(long)n * CC];
  ssum[threadIdx.x] = sum;
  __syncthreads();
  if (half == 0) G[b * CC + c] = (sum + ssum[threadIdx.x + 128]) * (1.f / NN);
}

// ---------------- dense head (tiny): one block per config ----------------
__global__ __launch_bounds__(128) void k_head(
    const float* __restrict__ g,
    const float* __restrict__ d1w, const float* __restrict__ d1b,
    const float* __restrict__ a1w1, const float* __restrict__ a1b1,
    const float* __restrict__ a1w2, const float* __restrict__ a1b2,
    const float* __restrict__ d2w, const float* __restrict__ d2b,
    const float* __restrict__ a2w1, const float* __restrict__ a2b1,
    const float* __restrict__ a2w2, const float* __restrict__ a2b2,
    const float* __restrict__ clsw, const float* __restrict__ clsb,
    float* __restrict__ out) {
  __shared__ float xv[CC];
  __shared__ float t[HH];
  __shared__ float red[CC];
  int b = blockIdx.x, c = threadIdx.x;
  // dense1
  xv[c] = g[b * CC + c];
  __syncthreads();
  float z = d1b[c];
  for (int k = 0; k < CC; ++k) z += xv[k] * d1w[k * CC + c];
  __syncthreads();
  xv[c] = z;
  __syncthreads();
  if (c < HH) {
    float s = a1b1[c];
    for (int k = 0; k < CC; ++k) s += xv[k] * a1w1[k * HH + c];
    t[c] = fmaxf(s, 0.f);
  }
  __syncthreads();
  float s2 = a1b2[c];
  for (int j = 0; j < HH; ++j) s2 += t[j] * a1w2[j * CC + c];
  float y = fmaxf(z / (1.f + __expf(-s2)), 0.f);
  __syncthreads();
  // dense2
  xv[c] = y;
  __syncthreads();
  float z2 = d2b[c];
  for (int k = 0; k < CC; ++k) z2 += xv[k] * d2w[k * CC + c];
  __syncthreads();
  xv[c] = z2;
  __syncthreads();
  if (c < HH) {
    float s = a2b1[c];
    for (int k = 0; k < CC; ++k) s += xv[k] * a2w1[k * HH + c];
    t[c] = fmaxf(s, 0.f);
  }
  __syncthreads();
  float s3 = a2b2[c];
  for (int j = 0; j < HH; ++j) s3 += t[j] * a2w2[j * CC + c];
  float y2 = fmaxf(z2 / (1.f + __expf(-s3)), 0.f);
  // classifier
  red[c] = y2 * clsw[c];
  __syncthreads();
  for (int st = 64; st > 0; st >>= 1) {
    if (c < st) red[c] += red[c + st];
    __syncthreads();
  }
  if (c == 0) out[b] = red[0] + clsb[0];
}

// ---------------- host launch ----------------
extern "C" void kernel_launch(void* const* d_in, const int* in_sizes, int n_in,
                              void* d_out, int out_size, void* d_ws, size_t ws_size,
                              hipStream_t stream) {
  (void)in_sizes; (void)n_in; (void)out_size; (void)ws_size;
  const int*   x_node_cfg = (const int*)d_in[0];
  const float* x_feat     = (const float*)d_in[1];
  const int*   x_lay      = (const int*)d_in[2];
  const int*   x_op       = (const int*)d_in[3];
  const int*   edge_index = (const int*)d_in[4];
  const int*   cfg_ids    = (const int*)d_in[5];
  const float* emb_op     = (const float*)d_in[6];
  const float* emb_cfg    = (const float*)d_in[7];
  const float* lin0_w = (const float*)d_in[8];
  const float* lin0_b = (const float*)d_in[9];
  const float* attn0_w1 = (const float*)d_in[10];
  const float* attn0_b1 = (const float*)d_in[11];
  const float* attn0_w2 = (const float*)d_in[12];
  const float* attn0_b2 = (const float*)d_in[13];
  const float* conv_wl = (const float*)d_in[14];
  const float* conv_bl = (const float*)d_in[15];
  const float* conv_wr = (const float*)d_in[16];
  const float* cattn_w1 = (const float*)d_in[17];
  const float* cattn_b1 = (const float*)d_in[18];
  const float* cattn_w2 = (const float*)d_in[19];
  const float* cattn_b2 = (const float*)d_in[20];
  const float* d1_w = (const float*)d_in[21];
  const float* d1_b = (const float*)d_in[22];
  const float* dattn1_w1 = (const float*)d_in[23];
  const float* dattn1_b1 = (const float*)d_in[24];
  const float* dattn1_w2 = (const float*)d_in[25];
  const float* dattn1_b2 = (const float*)d_in[26];
  const float* d2_w = (const float*)d_in[27];
  const float* d2_b = (const float*)d_in[28];
  const float* dattn2_w1 = (const float*)d_in[29];
  const float* dattn2_b1 = (const float*)d_in[30];
  const float* dattn2_w2 = (const float*)d_in[31];
  const float* dattn2_b2 = (const float*)d_in[32];
  const float* cls_w = (const float*)d_in[33];
  const float* cls_b = (const float*)d_in[34];
  float* out = (float*)d_out;

  // ---- workspace carve-up (Xcat bf16 aliases Agg f32: identical byte size) ----
  char* w = (char*)d_ws;
  size_t off = 0;
  uint16_t* Xcat = (uint16_t*)(w + off);
  float*    Agg  = (float*)(w + off);         off += (size_t)BN * KCAT * 2;  // == BN*CC*4
  float* Hbuf  = (float*)(w + off);           off += (size_t)BN * CC * 4;
  float* Ybuf  = (float*)(w + off);           off += (size_t)BN * CC * 4;
  float* stats = (float*)(w + off);           off += (size_t)BB * 2 * CC * 4;
  float* gbuf  = (float*)(w + off);           off += (size_t)BB * CC * 4;
  float* deg   = (float*)(w + off);           off += (size_t)NN * 4;
  int*   cfg_pos = (int*)(w + off);           off += (size_t)NN * 4;
  u32* P_lin0 = (u32*)(w + off);              off += (size_t)(KCAT / 32) * 8 * 32 * 8 * 4;
  u32* P_wl   = (u32*)(w + off);              off += (size_t)3 * (CC / 32) * 8 * 32 * 8 * 4;
  u32* P_wr   = (u32*)(w + off);              off += (size_t)3 * (CC / 32) * 8 * 32 * 8 * 4;

  const int packL = (KCAT / 32) * 8 * 32 * 8;  // 16384
  const int packC = (CC / 32) * 8 * 32 * 8;    // 8192

  // graph / map setup
  k_init_misc<<<(NN + 255) / 256, 256, 0, stream>>>(cfg_pos, deg);
  k_scatter_cfg<<<(NCC + 255) / 256, 256, 0, stream>>>(cfg_ids, cfg_pos);
  k_deg<<<(EE + 255) / 256, 256, 0, stream>>>(edge_index, deg);
  k_invdeg<<<(NN + 255) / 256, 256, 0, stream>>>(deg);

  // weight packing (bf16 WMMA B-operand images)
  k_pack_w<<<(packL + 255) / 256, 256, 0, stream>>>(lin0_w, P_lin0, 234, packL);
  for (int l = 0; l < 3; ++l) {
    k_pack_w<<<(packC + 255) / 256, 256, 0, stream>>>(conv_wl + (size_t)l * CC * CC,
                                                      P_wl + (size_t)l * packC, CC, packC);
    k_pack_w<<<(packC + 255) / 256, 256, 0, stream>>>(conv_wr + (size_t)l * CC * CC,
                                                      P_wr + (size_t)l * packC, CC, packC);
  }

  // feature concat -> bf16
  k_build_xcat<<<BN, 256, 0, stream>>>(x_node_cfg, x_feat, x_lay, x_op,
                                       cfg_pos, emb_op, emb_cfg, Xcat);

  // lin0: WMMA GEMM, then instnorm + attention + relu -> Hbuf
  k_gemm_lin0<<<BN / 16, 256, 0, stream>>>(Xcat, P_lin0, lin0_b, Ybuf);
  k_stats<<<BB, 256, 0, stream>>>(Ybuf, stats);
  k_norm_attn<<<BN, 128, 0, stream>>>(Ybuf, stats, nullptr,
                                      attn0_w1, attn0_b1, attn0_w2, attn0_b2,
                                      Hbuf, 0);

  // 3 SAGEConv layers
  for (int l = 0; l < 3; ++l) {
    k_zero<<<(int)(((long)BN * CC + 255) / 256), 256, 0, stream>>>(Agg, (long)BN * CC);
    k_scatter<<<EE, 256, 0, stream>>>(edge_index, deg, Hbuf, Agg);
    k_gemm_conv<<<BN / 16, 256, 0, stream>>>(Agg, Hbuf,
                                             P_wl + (size_t)l * packC,
                                             P_wr + (size_t)l * packC,
                                             conv_bl + (size_t)l * CC, Ybuf);
    k_stats<<<BB, 256, 0, stream>>>(Ybuf, stats);
    k_norm_attn<<<BN, 128, 0, stream>>>(Ybuf, stats, Hbuf,
                                        cattn_w1 + (size_t)l * CC * HH,
                                        cattn_b1 + (size_t)l * HH,
                                        cattn_w2 + (size_t)l * HH * CC,
                                        cattn_b2 + (size_t)l * CC,
                                        Hbuf, 1);
  }

  // pool + head
  k_pool<<<BB, 256, 0, stream>>>(Hbuf, gbuf);
  k_head<<<BB, 128, 0, stream>>>(gbuf,
                                 d1_w, d1_b, dattn1_w1, dattn1_b1, dattn1_w2, dattn1_b2,
                                 d2_w, d2_b, dattn2_w1, dattn2_b1, dattn2_w2, dattn2_b2,
                                 cls_w, cls_b, out);
}
